// SimpleGraphSageNet_52759378264356
// MI455X (gfx1250) — matrix operand order repaired and checked by reading, not compiled, not run on hardware
//
#include <hip/hip_runtime.h>

#define NN 100000
#define NE 1600000
#define HID 64
#define BUN 256
#define NSTRIPS (NN / 16)   // 6250, exact

typedef __attribute__((ext_vector_type(16))) __bf16 v16bf;
typedef __attribute__((ext_vector_type(8)))  __bf16 v8bf;
typedef __attribute__((ext_vector_type(8)))  float  v8f;

__device__ __forceinline__ __bf16 f2bf(float f) {
  unsigned u = __builtin_bit_cast(unsigned, f);
  unsigned r = u + 0x7FFFu + ((u >> 16) & 1u);   // round-to-nearest-even
  unsigned short hs = (unsigned short)(r >> 16);
  return __builtin_bit_cast(__bf16, hs);
}

// ---------------- utility kernels ----------------
__global__ void zero_f32(float* __restrict__ p, int n) {
  int t = blockIdx.x * blockDim.x + threadIdx.x;
  if (t < n) p[t] = 0.0f;
}

__global__ void cvt_bf16(const float* __restrict__ in, __bf16* __restrict__ out, int n) {
  int t = blockIdx.x * blockDim.x + threadIdx.x;
  if (t < n) out[t] = f2bf(in[t]);
}

// ---------------- degree / norm ----------------
__global__ void deg_kernel(const int* __restrict__ dst, float* __restrict__ deg) {
  int e = blockIdx.x * blockDim.x + threadIdx.x;
  if (e < NE) atomicAdd(&deg[dst[e]], 1.0f);
}

__global__ void norm_kernel(const float* __restrict__ deg, float* __restrict__ nrm,
                            float* __restrict__ inv) {
  int n = blockIdx.x * blockDim.x + threadIdx.x;
  if (n < NN) {
    float d = fmaxf(deg[n], 1.0f);
    nrm[n] = rsqrtf(d);
    inv[n] = 1.0f / d;
  }
}

// ---------------- embedding GEMM: h = x @ W_emb^T + b_emb  (bf16 WMMA, f32 acc) ----
// One wave per 16-row strip of h. A: 16x32 bf16 chunks of x; B: 32x16 chunks of W^T
// (contiguous runs of W rows since B[k][n] = W[n][k]).
__global__ void gemm_emb(const __bf16* __restrict__ xbf, const __bf16* __restrict__ wbf,
                         const float* __restrict__ bias, float* __restrict__ h) {
  const int lane  = threadIdx.x & 31;
  const int wave  = threadIdx.x >> 5;
  const int strip = blockIdx.x * 8 + wave;
  if (strip >= NSTRIPS) return;            // wave-uniform; EXEC all-1s inside
  const int row0 = strip * 16;
  const int m    = row0 + (lane & 15);     // A row handled by this lane
  const int hsel = lane >> 4;              // half-wave select

  // A k order chosen here (same as B): elem i -> K = hsel*16 + i
  const v8bf* a0lo = (const v8bf*)(xbf + (size_t)m * HID + 0  + hsel * 16);
  const v8bf* a0hi = (const v8bf*)(xbf + (size_t)m * HID + 8  + hsel * 16);
  const v8bf* a1lo = (const v8bf*)(xbf + (size_t)m * HID + 32 + hsel * 16);
  const v8bf* a1hi = (const v8bf*)(xbf + (size_t)m * HID + 40 + hsel * 16);
  v16bf a0 = __builtin_shufflevector(*a0lo, *a0hi, 0,1,2,3,4,5,6,7,8,9,10,11,12,13,14,15);
  v16bf a1 = __builtin_shufflevector(*a1lo, *a1hi, 0,1,2,3,4,5,6,7,8,9,10,11,12,13,14,15);

#pragma unroll
  for (int nt = 0; nt < 4; ++nt) {
    const int col = nt * 16 + (lane & 15);
    // B k order chosen here (same as A): elem i -> K = hsel*16 + i, N = lane&15; contiguous in W row.
    const v16bf* b0 = (const v16bf*)(wbf + (size_t)col * HID + 0  + hsel * 16);
    const v16bf* b1 = (const v16bf*)(wbf + (size_t)col * HID + 32 + hsel * 16);
    const float bv = bias[col];
    v8f acc;
#pragma unroll
    for (int i = 0; i < 8; ++i) acc[i] = bv;
    acc = __builtin_amdgcn_wmma_f32_16x16x32_bf16(false, a0, false, *b0, (short)0, acc, false, false);
    acc = __builtin_amdgcn_wmma_f32_16x16x32_bf16(false, a1, false, *b1, (short)0, acc, false, false);
    // C/D layout: VGPR v holds row M = v + 8*hsel, lane gives N
#pragma unroll
    for (int v = 0; v < 8; ++v)
      h[(size_t)(row0 + v + 8 * hsel) * HID + col] = acc[v];
  }
}

// ---------------- per-layer elementwise ----------------
__global__ void scale_rows(const float* __restrict__ h, const float* __restrict__ nrm,
                           float* __restrict__ hn, float* __restrict__ bcol0) {
  int t = blockIdx.x * blockDim.x + threadIdx.x;
  if (t >= NN * HID) return;
  int n = t >> 6;
  float v = h[t] * nrm[n];
  hn[t] = v;
  if (bcol0) bcol0[(size_t)n * BUN + (t & 63)] = v;   // layer 0: bundle cols [0,64)
}

__global__ void scatter_edges(const float* __restrict__ hn, const int* __restrict__ src,
                              const int* __restrict__ dst, float* __restrict__ c) {
  long long t = (long long)blockIdx.x * blockDim.x + threadIdx.x;
  if (t >= (long long)NE * HID) return;
  int e = (int)(t >> 6);
  int j = (int)(t & 63);
  float v = hn[(size_t)src[e] * HID + j];
  atomicAdd(&c[(size_t)dst[e] * HID + j], v);
}

__global__ void update_layer(const float* __restrict__ c, const float* __restrict__ inv,
                             const float* __restrict__ nrm, float* __restrict__ b,
                             float* __restrict__ h, int colOff) {
  int t = blockIdx.x * blockDim.x + threadIdx.x;
  if (t >= NN * HID) return;
  int n = t >> 6, j = t & 63;
  float cm = c[t] * inv[n];                 // mean aggregation
  b[(size_t)n * BUN + colOff + j] = cm;     // append to bundle
  h[t] += cm * nrm[n];                      // residual: h = h_in + c*norm
}

// Row L2-normalize first `width` cols of b; one wave32 per node.
__global__ void rownorm(float* __restrict__ b, int width) {
  int lane = threadIdx.x & 31;
  int node = (blockIdx.x * blockDim.x + threadIdx.x) >> 5;
  if (node >= NN) return;
  float* row = b + (size_t)node * BUN;
  float ss = 0.0f;
  for (int col = lane; col < width; col += 32) { float v = row[col]; ss += v * v; }
#pragma unroll
  for (int o = 16; o; o >>= 1) ss += __shfl_xor(ss, o, 32);
  float sc = 1.0f / fmaxf(sqrtf(ss), 1e-12f);
  for (int col = lane; col < width; col += 32) row[col] *= sc;
}

// ---------------- head: colsum(b) then tiny dense layers ----------------
__global__ void colsum(const float* __restrict__ b, float* __restrict__ s, int rowsPerBlock) {
  int j  = threadIdx.x;                  // 256 threads = 256 columns
  int r0 = blockIdx.x * rowsPerBlock;
  int r1 = r0 + rowsPerBlock; if (r1 > NN) r1 = NN;
  float acc = 0.0f;
  for (int r = r0; r < r1; ++r) {
    __builtin_prefetch(&b[(size_t)(r + 8) * BUN + j], 0, 0);  // global_prefetch_b8
    acc += b[(size_t)r * BUN + j];
  }
  atomicAdd(&s[j], acc);
}

__global__ void head_kernel(const float* __restrict__ s, const float* __restrict__ W_lin,
                            const float* __restrict__ b_lin, const float* __restrict__ W_out,
                            const float* __restrict__ b_out, float* __restrict__ out) {
  __shared__ float hg[HID];
  int j = threadIdx.x;                   // 64 threads
  float acc = 0.0f;
  for (int k = 0; k < BUN; ++k) acc += s[k] * W_lin[j * BUN + k];
  hg[j] = acc * (1.0f / (float)NN) + b_lin[j];   // mean ∘ linear == linear on colsum/N
  __syncthreads();
  if (j < 10) {
    float o = b_out[j];
    for (int k = 0; k < HID; ++k) o += hg[k] * W_out[j * HID + k];
    out[j] = o;
  }
}

// ---------------- host launcher ----------------
extern "C" void kernel_launch(void* const* d_in, const int* in_sizes, int n_in,
                              void* d_out, int out_size, void* d_ws, size_t ws_size,
                              hipStream_t stream) {
  const float* x     = (const float*)d_in[0];
  const int*   src   = (const int*)  d_in[1];
  const int*   dst   = (const int*)  d_in[2];
  const float* W_emb = (const float*)d_in[3];
  const float* b_emb = (const float*)d_in[4];
  const float* W_lin = (const float*)d_in[5];
  const float* b_lin = (const float*)d_in[6];
  const float* W_out = (const float*)d_in[7];
  const float* b_out = (const float*)d_in[8];
  float* out = (float*)d_out;

  char* base = (char*)d_ws;
  size_t off = 0;
  auto take = [&](size_t bytes) -> char* {
    char* p = base + off;
    off += (bytes + 255) & ~(size_t)255;
    return p;
  };
  float*  h    = (float*) take((size_t)NN * HID * 4);
  float*  hn   = (float*) take((size_t)NN * HID * 4);
  float*  c    = (float*) take((size_t)NN * HID * 4);
  float*  b    = (float*) take((size_t)NN * BUN * 4);
  float*  deg  = (float*) take((size_t)NN * 4);
  float*  nrm  = (float*) take((size_t)NN * 4);
  float*  inv  = (float*) take((size_t)NN * 4);
  float*  s    = (float*) take(BUN * 4);
  __bf16* xbf  = (__bf16*)take((size_t)NN * HID * 2);
  __bf16* wbf  = (__bf16*)take((size_t)HID * HID * 2);

  const int TPB = 256;
  const int NH_BLK = (NN * HID + TPB - 1) / TPB;   // 25000

  // init / precision conversion
  zero_f32<<<(NN + TPB - 1) / TPB, TPB, 0, stream>>>(deg, NN);
  zero_f32<<<1, TPB, 0, stream>>>(s, BUN);
  cvt_bf16<<<NH_BLK, TPB, 0, stream>>>(x, xbf, NN * HID);
  cvt_bf16<<<(HID * HID + TPB - 1) / TPB, TPB, 0, stream>>>(W_emb, wbf, HID * HID);

  // degrees -> norm / inv_deg
  deg_kernel<<<(NE + TPB - 1) / TPB, TPB, 0, stream>>>(dst, deg);
  norm_kernel<<<(NN + TPB - 1) / TPB, TPB, 0, stream>>>(deg, nrm, inv);

  // embedding GEMM (WMMA bf16, f32 accumulate)
  gemm_emb<<<(NSTRIPS + 7) / 8, TPB, 0, stream>>>(xbf, wbf, b_emb, h);

  // 3 message-passing layers
  for (int l = 0; l < 3; ++l) {
    scale_rows<<<NH_BLK, TPB, 0, stream>>>(h, nrm, hn, (l == 0) ? b : nullptr);
    zero_f32<<<NH_BLK, TPB, 0, stream>>>(c, NN * HID);
    long long tot = (long long)NE * HID;
    scatter_edges<<<(int)((tot + TPB - 1) / TPB), TPB, 0, stream>>>(hn, src, dst, c);
    update_layer<<<NH_BLK, TPB, 0, stream>>>(c, inv, nrm, b, h, 64 * (l + 1));
    rownorm<<<(NN * 32 + TPB - 1) / TPB, TPB, 0, stream>>>(b, 64 * (l + 2));
  }

  // head: collapse (b @ W_lin^T -> mean) into colsum(b)/N @ W_lin^T
  colsum<<<400, TPB, 0, stream>>>(b, s, 250);
  head_kernel<<<1, 64, 0, stream>>>(s, W_lin, b_lin, W_out, b_out, out);
}
